// TransformerAli_73581379715525
// MI455X (gfx1250) — matrix operand order repaired and checked by reading, not compiled, not run on hardware
//
#include <hip/hip_runtime.h>
#include <stdint.h>

// ---------------------------------------------------------------------------
// Problem constants (from reference)
// ---------------------------------------------------------------------------
#define SEQ   1024
#define BATCH 4
#define EMB   768
#define NHEAD 12
#define HDIM  64
#define NLAYER 4
#define MROWS (SEQ * BATCH)      // 4096
#define E3    (3 * EMB)          // 2304
#define E4    (4 * EMB)          // 3072
#define BHEAD (BATCH * NHEAD)    // 48

typedef __attribute__((ext_vector_type(16))) __bf16 v16bf;
typedef __attribute__((ext_vector_type(8)))  float  v8f;

union FragAB { v16bf v; uint4 q[2]; unsigned short u[16]; };
union FragC  { v8f   v; float f[8]; };
union Pack8  { uint4 q; unsigned short u[8]; };

__device__ __forceinline__ unsigned short f2bf(float x) {
  unsigned int u = __builtin_bit_cast(unsigned int, x);
  u += 0x7fffu + ((u >> 16) & 1u);   // round-to-nearest-even
  return (unsigned short)(u >> 16);
}

__device__ __forceinline__ float fast_rcp(float x) {
  return __builtin_amdgcn_rcpf(x);   // single v_rcp_f32
}
__device__ __forceinline__ float fast_sigmoid(float x) {
  return fast_rcp(1.0f + __expf(-x));
}

// ---------------------------------------------------------------------------
// fp32 -> bf16 bulk convert (weights)
// ---------------------------------------------------------------------------
__global__ void cvt_f32_bf16(const float* __restrict__ src,
                             unsigned short* __restrict__ dst, size_t n) {
  size_t i = (size_t)blockIdx.x * blockDim.x + threadIdx.x;
  size_t stride = (size_t)gridDim.x * blockDim.x;
  for (; i < n; i += stride) dst[i] = f2bf(src[i]);
}

// ---------------------------------------------------------------------------
// LayerNorm over E=768, one block per row, emits bf16 row
// ---------------------------------------------------------------------------
__global__ __launch_bounds__(256)
void ln_bf16_kernel(const float* __restrict__ x, const float* __restrict__ g,
                    const float* __restrict__ bta, unsigned short* __restrict__ out) {
  const int row  = blockIdx.x;
  const int tid  = threadIdx.x;
  const float* xr = x + (size_t)row * EMB;
  float v0 = xr[tid], v1 = xr[tid + 256], v2 = xr[tid + 512];
  float s  = v0 + v1 + v2;
  float ss = v0 * v0 + v1 * v1 + v2 * v2;
  #pragma unroll
  for (int m = 16; m >= 1; m >>= 1) {
    s  += __shfl_xor(s,  m);
    ss += __shfl_xor(ss, m);
  }
  __shared__ float rs[8], rss[8];
  const int wid = tid >> 5, lane = tid & 31;
  if (lane == 0) { rs[wid] = s; rss[wid] = ss; }
  __syncthreads();
  float S = 0.f, SS = 0.f;
  #pragma unroll
  for (int i = 0; i < 8; i++) { S += rs[i]; SS += rss[i]; }
  const float mu   = S * (1.0f / EMB);
  const float var  = SS * (1.0f / EMB) - mu * mu;
  const float rstd = rsqrtf(var + 1e-5f);
  unsigned short* orow = out + (size_t)row * EMB;
  orow[tid]       = f2bf((v0 - mu) * rstd * g[tid]       + bta[tid]);
  orow[tid + 256] = f2bf((v1 - mu) * rstd * g[tid + 256] + bta[tid + 256]);
  orow[tid + 512] = f2bf((v2 - mu) * rstd * g[tid + 512] + bta[tid + 512]);
}

// ---------------------------------------------------------------------------
// bf16 WMMA GEMM: C[M,N] = A[M,K] * W[N,K]^T (+bias, + fused epilogue)
// 128x64 block tile, 256 threads = 8 waves, each wave 32x32 (4 WMMA / k-step)
// K stepped by 32 through double-buffered LDS (register-staged prefetch).
// MODE 0: qkv scatter   MODE 1: fp32 residual add   MODE 2: QuickGELU -> bf16
// ---------------------------------------------------------------------------
template <int MODE>
__global__ __launch_bounds__(256)
void gemm_bf16_wmma(const unsigned short* __restrict__ A,
                    const unsigned short* __restrict__ W,
                    const float* __restrict__ bias,
                    float* __restrict__ xresid,
                    unsigned short* __restrict__ obf,
                    unsigned short* __restrict__ qb,
                    unsigned short* __restrict__ kb,
                    unsigned short* __restrict__ vb,
                    int Ndim, int Kdim) {
  __shared__ __align__(16) unsigned short Atile[2][128 * 32];  // 16 KB
  __shared__ __align__(16) unsigned short Btile[2][64 * 32];   //  8 KB

  const int m0   = blockIdx.y * 128;
  const int n0   = blockIdx.x * 64;
  const int tid  = threadIdx.x;
  const int lane = tid & 31;
  const int wid  = tid >> 5;          // 0..7
  const int wm   = (wid >> 1) * 32;   // 0,32,64,96
  const int wn   = (wid & 1) * 32;    // 0,32
  const int laneLo = lane & 15;
  const int laneHi = lane >> 4;
  const int hi8    = laneHi * 8;

  // loader indices
  const int aRow = tid >> 1, aCol = (tid & 1) * 16;  // 128 x 32, 16 elems/thread
  const int bRow = tid >> 2, bCol = (tid & 3) * 8;   //  64 x 32,  8 elems/thread

  uint4 ra0, ra1, rb0;
  auto loadRegs = [&](int k0) {
    const uint4* ga = (const uint4*)(A + (size_t)(m0 + aRow) * Kdim + k0 + aCol);
    ra0 = ga[0]; ra1 = ga[1];
    rb0 = *(const uint4*)(W + (size_t)(n0 + bRow) * Kdim + k0 + bCol);
  };
  auto storeLds = [&](int buf) {
    uint4* la = (uint4*)(Atile[buf] + aRow * 32 + aCol);
    la[0] = ra0; la[1] = ra1;
    *(uint4*)(Btile[buf] + bRow * 32 + bCol) = rb0;
  };

  FragC acc[2][2];
  #pragma unroll
  for (int i = 0; i < 2; i++)
    #pragma unroll
    for (int j = 0; j < 2; j++)
      #pragma unroll
      for (int r = 0; r < 8; r++) acc[i][j].f[r] = 0.0f;

  loadRegs(0);
  storeLds(0);
  __syncthreads();
  int cur = 0;

  for (int k0 = 0; k0 < Kdim; k0 += 32) {
    const bool hasNext = (k0 + 32) < Kdim;
    if (hasNext) loadRegs(k0 + 32);   // global loads overlap the WMMAs below

    FragAB a[2], b[2];
    #pragma unroll
    for (int t = 0; t < 2; t++) {
      // A fragment: lane M = laneLo; elems 0..7 <-> K = hi8+e, 8..15 <-> 16+hi8+e
      const unsigned short* ap = Atile[cur] + (wm + t * 16 + laneLo) * 32;
      a[t].q[0] = *(const uint4*)(ap + hi8);
      a[t].q[1] = *(const uint4*)(ap + 16 + hi8);
      // B fragment: lane N = laneLo; elems 0..7 <-> K = hi8+e, 8..15 <-> 16+hi8+e
      const unsigned short* bp = Btile[cur] + (wn + t * 16 + laneLo) * 32;
      b[t].q[0] = *(const uint4*)(bp + hi8);
      b[t].q[1] = *(const uint4*)(bp + 16 + hi8);
    }
    #pragma unroll
    for (int i = 0; i < 2; i++)
      #pragma unroll
      for (int j = 0; j < 2; j++)
        acc[i][j].v = __builtin_amdgcn_wmma_f32_16x16x32_bf16(
            false, a[i].v, false, b[j].v, (short)0, acc[i][j].v, false, false);

    if (hasNext) {
      __syncthreads();
      storeLds(cur ^ 1);
      __syncthreads();
      cur ^= 1;
    }
  }

  // ------------------ epilogue ------------------
  #pragma unroll
  for (int i = 0; i < 2; i++) {
    #pragma unroll
    for (int j = 0; j < 2; j++) {
      #pragma unroll
      for (int r = 0; r < 8; r++) {
        const int m = m0 + wm + i * 16 + r + 8 * laneHi;
        const int n = n0 + wn + j * 16 + laneLo;
        float val = acc[i][j].f[r] + bias[n];
        if (MODE == 0) {
          // scatter into q/k/v [bh, s, d]; q scaled by HD^-0.5
          const int which = n / EMB;
          const int rem   = n % EMB;
          const int h = rem >> 6, d = rem & 63;
          const int s = m >> 2, bb = m & 3;
          const size_t idx = ((size_t)(bb * NHEAD + h) * SEQ + s) * HDIM + d;
          if (which == 0)      qb[idx] = f2bf(val * 0.125f);
          else if (which == 1) kb[idx] = f2bf(val);
          else                 vb[idx] = f2bf(val);
        } else if (MODE == 1) {
          xresid[(size_t)m * EMB + n] += val;                     // residual add
        } else {
          const float gl = val * fast_sigmoid(1.702f * val);      // QuickGELU
          obf[(size_t)m * Ndim + n] = f2bf(gl);
        }
      }
    }
  }
}

// ---------------------------------------------------------------------------
// Flash-style WMMA attention. Block = 128 threads = 4 waves; each wave owns a
// 16-query tile; K/V (32 keys x 64) double-buffered in LDS; online softmax;
// P transposed through per-wave LDS into the A-fragment layout.
// ALiBi smoothing sigmoid is hoisted: sw(max(m,t)) == sw(m) when m>=t (8
// loop-invariant values) else sw(t) (2 values per key block).
// ---------------------------------------------------------------------------
__global__ __launch_bounds__(128)
void attn_wmma(const unsigned short* __restrict__ q,
               const unsigned short* __restrict__ k,
               const unsigned short* __restrict__ v,
               unsigned short* __restrict__ o) {
  __shared__ __align__(16) unsigned short Kt[2][32 * 64];    // [key][d]   8 KB
  __shared__ __align__(16) unsigned short Vt[2][64 * 32];    // [d][key]   8 KB
  __shared__ __align__(16) unsigned short Pbuf[4][16 * 32];  // per-wave   4 KB

  const int bh  = blockIdx.y;
  const int bb  = bh / NHEAD;
  const int h   = bh % NHEAD;
  const int hh  = bh >> 2;              // faithful repeat_interleave head index
  const int tid = threadIdx.x;
  const int lane = tid & 31;
  const int wid  = tid >> 5;
  const int laneLo = lane & 15;
  const int laneHi = lane >> 4;
  const int hi8    = laneHi * 8;
  const int qs0 = (blockIdx.x * 4 + wid) * 16;

  const float balpha = -0.1f * exp2f(-(float)hh * (1.0f / 12.0f));  // -alpha*SCALE

  // K/V tile loaders (register-staged, double-buffered LDS)
  const int kKey = tid >> 2, kDb = (tid & 3) * 16;        // K: 16 elems/thread
  const int vKey = (tid & 15) * 2, vDb = (tid >> 4) * 8;  // V: 2 keys x 8 d
  uint4 rk0, rk1;
  Pack8 rv0, rv1;
  auto loadKV = [&](int kb0) {
    const uint4* ks = (const uint4*)(k + ((size_t)bh * SEQ + kb0 + kKey) * HDIM + kDb);
    rk0 = ks[0]; rk1 = ks[1];
    rv0.q = *(const uint4*)(v + ((size_t)bh * SEQ + kb0 + vKey)     * HDIM + vDb);
    rv1.q = *(const uint4*)(v + ((size_t)bh * SEQ + kb0 + vKey + 1) * HDIM + vDb);
  };
  auto storeKV = [&](int buf) {
    uint4* kd = (uint4*)(Kt[buf] + kKey * 64 + kDb);
    kd[0] = rk0; kd[1] = rk1;
    #pragma unroll
    for (int j = 0; j < 8; j++) {   // packed transpose: two keys per b32 store
      unsigned int w = (unsigned int)rv0.u[j] | ((unsigned int)rv1.u[j] << 16);
      *(unsigned int*)(Vt[buf] + (vDb + j) * 32 + vKey) = w;
    }
  };

  // Q fragments (reused for all key blocks): A-layout over d = 0..63
  FragAB qf[2];
  {
    const unsigned short* qrow = q + ((size_t)bh * SEQ + qs0 + laneLo) * HDIM;
    #pragma unroll
    for (int f = 0; f < 2; f++) {
      qf[f].q[0] = *(const uint4*)(qrow + f * 32 + hi8);
      qf[f].q[1] = *(const uint4*)(qrow + f * 32 + 16 + hi8);
    }
  }

  FragC oacc[4];
  float rowmax[8], rowsum[8];
  float mf[8], swm[8];   // query positions + their (loop-invariant) sigmoids
  #pragma unroll
  for (int nb = 0; nb < 4; nb++)
    #pragma unroll
    for (int r = 0; r < 8; r++) oacc[nb].f[r] = 0.0f;
  #pragma unroll
  for (int r = 0; r < 8; r++) {
    rowmax[r] = -1e30f;
    rowsum[r] = 0.0f;
    mf[r]  = (float)(qs0 + r + 8 * laneHi);
    swm[r] = fast_sigmoid(0.5f * (mf[r] - 20.0f));
  }

  loadKV(0);
  storeKV(0);
  __syncthreads();
  int cur = 0;

  for (int kb0 = 0; kb0 < SEQ; kb0 += 32) {
    const bool hasNext = (kb0 + 32) < SEQ;
    if (hasNext) loadKV(kb0 + 32);   // overlap with compute below

    // scores: two 16x16 tiles over this 32-key block
    FragC sc[2];
    #pragma unroll
    for (int tn = 0; tn < 2; tn++) {
      FragAB kf0, kf1;  // B-layout: N = key, K = d
      const unsigned short* kp = Kt[cur] + (tn * 16 + laneLo) * 64 + hi8;
      kf0.q[0] = *(const uint4*)(kp);      kf0.q[1] = *(const uint4*)(kp + 16);
      kf1.q[0] = *(const uint4*)(kp + 32); kf1.q[1] = *(const uint4*)(kp + 48);
      FragC z;
      #pragma unroll
      for (int r = 0; r < 8; r++) z.f[r] = 0.0f;
      z.v = __builtin_amdgcn_wmma_f32_16x16x32_bf16(false, qf[0].v, false, kf0.v,
                                                    (short)0, z.v, false, false);
      z.v = __builtin_amdgcn_wmma_f32_16x16x32_bf16(false, qf[1].v, false, kf1.v,
                                                    (short)0, z.v, false, false);
      sc[tn] = z;
    }

    // key-side sigmoids: 2 transcendental pairs per 32-key block (was 16)
    const float tf0  = (float)(kb0 + laneLo);
    const float tf1  = tf0 + 16.0f;
    const float swt0 = fast_sigmoid(0.5f * (tf0 - 20.0f));
    const float swt1 = fast_sigmoid(0.5f * (tf1 - 20.0f));

    // online softmax over the 16x32 block
    float p0[8], p1[8];
    #pragma unroll
    for (int r = 0; r < 8; r++) {
      const float d0 = mf[r] - tf0;
      const float d1 = mf[r] - tf1;
      const float sw0 = (d0 >= 0.0f) ? swm[r] : swt0;   // sw(max(m,t))
      const float sw1 = (d1 >= 0.0f) ? swm[r] : swt1;
      const float s0 = sc[0].f[r] + balpha * fabsf(d0) * sw0;
      const float s1 = sc[1].f[r] + balpha * fabsf(d1) * sw1;
      float bm = fmaxf(s0, s1);
      bm = fmaxf(bm, __shfl_xor(bm, 1));
      bm = fmaxf(bm, __shfl_xor(bm, 2));
      bm = fmaxf(bm, __shfl_xor(bm, 4));
      bm = fmaxf(bm, __shfl_xor(bm, 8));
      const float nm   = fmaxf(rowmax[r], bm);
      const float corr = __expf(rowmax[r] - nm);
      rowmax[r] = nm;
      const float e0 = __expf(s0 - nm), e1 = __expf(s1 - nm);
      float ps = e0 + e1;
      ps += __shfl_xor(ps, 1);
      ps += __shfl_xor(ps, 2);
      ps += __shfl_xor(ps, 4);
      ps += __shfl_xor(ps, 8);
      rowsum[r] = rowsum[r] * corr + ps;
      p0[r] = e0; p1[r] = e1;
      #pragma unroll
      for (int nb = 0; nb < 4; nb++) oacc[nb].f[r] *= corr;
    }

    // P -> per-wave LDS -> A fragment (layout transpose)
    unsigned short* pw = Pbuf[wid];
    #pragma unroll
    for (int r = 0; r < 8; r++) {
      const int row = r + 8 * laneHi;
      pw[row * 32 + laneLo]      = f2bf(p0[r]);
      pw[row * 32 + 16 + laneLo] = f2bf(p1[r]);
    }
    FragAB pf;
    {
      const unsigned short* pr = pw + laneLo * 32;
      pf.q[0] = *(const uint4*)(pr + hi8);
      pf.q[1] = *(const uint4*)(pr + 16 + hi8);
    }

    // O += P @ V   (4 d-subtiles of 16)
    #pragma unroll
    for (int nb = 0; nb < 4; nb++) {
      FragAB vf;  // B-layout: N = d, K = key (from transposed V tile)
      const unsigned short* vp = Vt[cur] + (nb * 16 + laneLo) * 32 + hi8;
      vf.q[0] = *(const uint4*)(vp);
      vf.q[1] = *(const uint4*)(vp + 16);
      oacc[nb].v = __builtin_amdgcn_wmma_f32_16x16x32_bf16(
          false, pf.v, false, vf.v, (short)0, oacc[nb].v, false, false);
    }

    if (hasNext) {
      __syncthreads();
      storeKV(cur ^ 1);
      __syncthreads();
      cur ^= 1;
    }
  }

  // epilogue: o[s, b, h*64+d] bf16, rows m = s*B + b
  float rinv[8];
  #pragma unroll
  for (int r = 0; r < 8; r++) rinv[r] = fast_rcp(rowsum[r]);
  #pragma unroll
  for (int nb = 0; nb < 4; nb++) {
    #pragma unroll
    for (int r = 0; r < 8; r++) {
      const int s_ = qs0 + r + 8 * laneHi;
      const int d  = nb * 16 + laneLo;
      o[((size_t)(s_ * BATCH + bb)) * EMB + h * HDIM + d] = f2bf(oacc[nb].f[r] * rinv[r]);
    }
  }
}

// ---------------------------------------------------------------------------
// Host launcher
// ---------------------------------------------------------------------------
extern "C" void kernel_launch(void* const* d_in, const int* in_sizes, int n_in,
                              void* d_out, int out_size, void* d_ws, size_t ws_size,
                              hipStream_t stream) {
  (void)in_sizes; (void)n_in; (void)out_size; (void)ws_size;

  const float* x_in    = (const float*)d_in[0];
  const float* w_qkv_f = (const float*)d_in[1];
  const float* b_qkv   = (const float*)d_in[2];
  const float* w_out_f = (const float*)d_in[3];
  const float* b_out   = (const float*)d_in[4];
  const float* ln1g    = (const float*)d_in[5];
  const float* ln1b    = (const float*)d_in[6];
  const float* w_fc_f  = (const float*)d_in[7];
  const float* b_fc    = (const float*)d_in[8];
  const float* w_pr_f  = (const float*)d_in[9];
  const float* b_pr    = (const float*)d_in[10];
  const float* ln2g    = (const float*)d_in[11];
  const float* ln2b    = (const float*)d_in[12];

  float* xbuf = (float*)d_out;   // working residual stream (also final output)

  char* ws = (char*)d_ws;
  size_t off = 0;
  auto alloc_us = [&](size_t nelem) -> unsigned short* {
    unsigned short* p = (unsigned short*)(ws + off);
    off = (off + nelem * 2 + 255) & ~(size_t)255;
    return p;
  };

  const size_t n_wqkv = (size_t)NLAYER * E3 * EMB;
  const size_t n_wout = (size_t)NLAYER * EMB * EMB;
  const size_t n_wfc  = (size_t)NLAYER * E4 * EMB;
  const size_t n_wpr  = (size_t)NLAYER * EMB * E4;

  unsigned short* w_qkv = alloc_us(n_wqkv);
  unsigned short* w_o   = alloc_us(n_wout);
  unsigned short* w_fc  = alloc_us(n_wfc);
  unsigned short* w_pr  = alloc_us(n_wpr);
  unsigned short* hbuf  = alloc_us((size_t)MROWS * EMB);
  unsigned short* qbuf  = alloc_us((size_t)BHEAD * SEQ * HDIM);
  unsigned short* kbuf  = alloc_us((size_t)BHEAD * SEQ * HDIM);
  unsigned short* vbuf  = alloc_us((size_t)BHEAD * SEQ * HDIM);
  unsigned short* obuf  = alloc_us((size_t)MROWS * EMB);
  unsigned short* abuf  = alloc_us((size_t)MROWS * E4);

  hipMemcpyAsync(xbuf, x_in, (size_t)MROWS * EMB * sizeof(float),
                 hipMemcpyDeviceToDevice, stream);

  cvt_f32_bf16<<<2048, 256, 0, stream>>>(w_qkv_f, w_qkv, n_wqkv);
  cvt_f32_bf16<<<2048, 256, 0, stream>>>(w_out_f, w_o,   n_wout);
  cvt_f32_bf16<<<2048, 256, 0, stream>>>(w_fc_f,  w_fc,  n_wfc);
  cvt_f32_bf16<<<2048, 256, 0, stream>>>(w_pr_f,  w_pr,  n_wpr);

  for (int l = 0; l < NLAYER; l++) {
    ln_bf16_kernel<<<MROWS, 256, 0, stream>>>(xbuf, ln1g + l * EMB, ln1b + l * EMB, hbuf);

    gemm_bf16_wmma<0><<<dim3(E3 / 64, MROWS / 128), 256, 0, stream>>>(
        hbuf, w_qkv + (size_t)l * E3 * EMB, b_qkv + (size_t)l * E3,
        nullptr, nullptr, qbuf, kbuf, vbuf, E3, EMB);

    attn_wmma<<<dim3(SEQ / 64, BHEAD), 128, 0, stream>>>(qbuf, kbuf, vbuf, obuf);

    gemm_bf16_wmma<1><<<dim3(EMB / 64, MROWS / 128), 256, 0, stream>>>(
        obuf, w_o + (size_t)l * EMB * EMB, b_out + (size_t)l * EMB,
        xbuf, nullptr, nullptr, nullptr, nullptr, EMB, EMB);

    ln_bf16_kernel<<<MROWS, 256, 0, stream>>>(xbuf, ln2g + l * EMB, ln2b + l * EMB, hbuf);

    gemm_bf16_wmma<2><<<dim3(E4 / 64, MROWS / 128), 256, 0, stream>>>(
        hbuf, w_fc + (size_t)l * E4 * EMB, b_fc + (size_t)l * E4,
        nullptr, abuf, nullptr, nullptr, nullptr, E4, EMB);

    gemm_bf16_wmma<1><<<dim3(EMB / 64, MROWS / 128), 256, 0, stream>>>(
        abuf, w_pr + (size_t)l * EMB * E4, b_pr + (size_t)l * EMB,
        xbuf, nullptr, nullptr, nullptr, nullptr, EMB, E4);
  }
}
